// ASMR_13159779795108
// MI455X (gfx1250) — matrix operand order repaired and checked by reading, not compiled, not run on hardware
//
#include <hip/hip_runtime.h>
#include <math.h>

#define N_ROWS 8192
#define D_TEXT 256
#define D_CAP  128
#define TILE   128
#define GRID_T (N_ROWS / TILE)     // 64
#define NPART  (GRID_T * GRID_T)   // 4096

typedef __attribute__((ext_vector_type(16))) __bf16        v16bf;
typedef __attribute__((ext_vector_type(8)))  float         v8f;
typedef __attribute__((ext_vector_type(4)))  unsigned int  u32x4;

union Frag {
  v16bf v;
  u32x4 u[2];
};

// A fragment (16x32 bf16, MxK). Layout (ISA 7.12.2): lanes 0-15 hold M=lane,
// K=0..7 in VGPR0-3 and K=16..23 in VGPR4-7; lanes 16-31 hold K=8..15 / K=24..31.
__device__ __forceinline__ v16bf load_frag_a(const __bf16* base, int ld,
                                             int row0, int k0, int lane) {
  int r = lane & 15, h = lane >> 4;
  const __bf16* p = base + (size_t)(row0 + r) * ld + (k0 + h * 8);
  Frag f;
  f.u[0] = *reinterpret_cast<const u32x4*>(p);        // K local h*8   .. h*8+7
  f.u[1] = *reinterpret_cast<const u32x4*>(p + 16);   // K local 16+h*8 ..
  return f.v;
}

// B fragment (32x16 bf16, KxN) where B = T^T of a row-major matrix:
// lane 0-15 holds column N=lane, K=k0..k0+7 and k0+16..k0+23;
// lanes 16-31 hold K=k0+8..k0+15 and k0+24..k0+31.
__device__ __forceinline__ v16bf load_frag_b(const __bf16* base, int ld,
                                             int col0, int k0, int lane) {
  int r = lane & 15, h = lane >> 4;
  const __bf16* p = base + (size_t)(col0 + r) * ld + (k0 + h * 8);
  Frag f;
  f.u[0] = *reinterpret_cast<const u32x4*>(p);        // K local h*8    .. h*8+7
  f.u[1] = *reinterpret_cast<const u32x4*>(p + 16);   // K local 16+h*8 ..
  return f.v;
}

// ---------------- prep kernels ----------------

// L2-normalize each text row (matches F.normalize eps), convert to bf16.
__global__ void prep_text_kernel(const float* __restrict__ x,
                                 __bf16* __restrict__ t) {
  __shared__ float sm[D_TEXT];
  int row = blockIdx.x, tid = threadIdx.x;
  float v = x[(size_t)row * D_TEXT + tid];
  sm[tid] = v * v;
  __syncthreads();
  for (int s = D_TEXT / 2; s > 0; s >>= 1) {
    if (tid < s) sm[tid] += sm[tid + s];
    __syncthreads();
  }
  float norm  = sqrtf(sm[0]);
  float scale = 1.0f / fmaxf(norm, 1e-12f);
  t[(size_t)row * D_TEXT + tid] = (__bf16)(v * scale);
}

// caption row squared norms (fp32) + bf16 copy for the GEMM.
__global__ void prep_cap_kernel(const float* __restrict__ x,
                                __bf16* __restrict__ c,
                                float* __restrict__ sq) {
  __shared__ float sm[D_CAP];
  int row = blockIdx.x, tid = threadIdx.x;
  float v = x[(size_t)row * D_CAP + tid];
  sm[tid] = v * v;
  __syncthreads();
  for (int s = D_CAP / 2; s > 0; s >>= 1) {
    if (tid < s) sm[tid] += sm[tid + s];
    __syncthreads();
  }
  if (tid == 0) sq[row] = sm[0];
  c[(size_t)row * D_CAP + tid] = (__bf16)v;
}

// mean(t t^T) == ||sum_n t_n||^2 / N^2 : one column-sum pass, deterministic.
__global__ void mean_sim_kernel(const __bf16* __restrict__ t,
                                float* __restrict__ msim) {
  __shared__ float sm[D_TEXT];
  int d = threadIdx.x;
  float s = 0.0f;
  for (int n = 0; n < N_ROWS; ++n) s += (float)t[(size_t)n * D_TEXT + d];
  sm[d] = s * s;
  __syncthreads();
  for (int st = D_TEXT / 2; st > 0; st >>= 1) {
    if (d < st) sm[d] += sm[d + st];
    __syncthreads();
  }
  if (d == 0) msim[0] = sm[0] * (1.0f / ((float)N_ROWS * (float)N_ROWS));
}

// ---------------- fused tile kernel ----------------
// Block = 256 threads = 8 waves covering a 128x128 output tile.
// Wave (wm 0..3, wn 0..1) owns a 32x64 sub-tile = 2x4 fragments of 16x16.
// Both GEMMs accumulated with v_wmma_f32_16x16x32_bf16, then the sigmoid /
// squared-error epilogue runs entirely in registers; tile partial -> ws.
__global__ void __launch_bounds__(256)
fused_tile_kernel(const __bf16* __restrict__ t,
                  const __bf16* __restrict__ c,
                  const float* __restrict__ sq,
                  const float* __restrict__ msim,
                  float* __restrict__ partials) {
  __shared__ float red[256];

  const int lane = threadIdx.x & 31;
  const int wave = threadIdx.x >> 5;
  const int wm = wave >> 1;   // 0..3
  const int wn = wave & 1;    // 0..1
  const int rbase = blockIdx.y * TILE + wm * 32;
  const int cbase = blockIdx.x * TILE + wn * 64;

  const v8f vzero = {0.f, 0.f, 0.f, 0.f, 0.f, 0.f, 0.f, 0.f};
  v8f acc_t[2][4];
  v8f acc_c[2][4];
#pragma unroll
  for (int i = 0; i < 2; ++i)
#pragma unroll
    for (int j = 0; j < 4; ++j) { acc_t[i][j] = vzero; acc_c[i][j] = vzero; }

  // ---- cosine_text partial: t[rows] . t[cols]^T, K = 256 ----
#pragma unroll
  for (int k0 = 0; k0 < D_TEXT; k0 += 32) {
    v16bf a[2], b[4];
#pragma unroll
    for (int i = 0; i < 2; ++i) a[i] = load_frag_a(t, D_TEXT, rbase + 16 * i, k0, lane);
#pragma unroll
    for (int j = 0; j < 4; ++j) b[j] = load_frag_b(t, D_TEXT, cbase + 16 * j, k0, lane);
#pragma unroll
    for (int i = 0; i < 2; ++i)
#pragma unroll
      for (int j = 0; j < 4; ++j)
        acc_t[i][j] = __builtin_amdgcn_wmma_f32_16x16x32_bf16(
            false, a[i], false, b[j], (short)0, acc_t[i][j], false, false);
  }

  // ---- caption gram partial: caption[rows] . caption[cols]^T, K = 128 ----
#pragma unroll
  for (int k0 = 0; k0 < D_CAP; k0 += 32) {
    v16bf a[2], b[4];
#pragma unroll
    for (int i = 0; i < 2; ++i) a[i] = load_frag_a(c, D_CAP, rbase + 16 * i, k0, lane);
#pragma unroll
    for (int j = 0; j < 4; ++j) b[j] = load_frag_b(c, D_CAP, cbase + 16 * j, k0, lane);
#pragma unroll
    for (int i = 0; i < 2; ++i)
#pragma unroll
      for (int j = 0; j < 4; ++j)
        acc_c[i][j] = __builtin_amdgcn_wmma_f32_16x16x32_bf16(
            false, a[i], false, b[j], (short)0, acc_c[i][j], false, false);
  }

  // ---- fused epilogue ----
  // C/D layout: element (vgpr v, lane) -> row = base + v + (lane>>4)*8,
  //                                       col = base + (lane&15)
  const int lr = lane & 15, lh = lane >> 4;
  const float ms = msim[0];
  float sqc[4];
#pragma unroll
  for (int j = 0; j < 4; ++j) sqc[j] = sq[cbase + 16 * j + lr];

  float local = 0.0f;
#pragma unroll
  for (int i = 0; i < 2; ++i) {
#pragma unroll
    for (int v = 0; v < 8; ++v) {
      float sqr = sq[rbase + 16 * i + lh * 8 + v];
#pragma unroll
      for (int j = 0; j < 4; ++j) {
        float ct   = acc_t[i][j][v];
        float cc   = acc_c[i][j][v];
        float dist = sqr + sqc[j] - 2.0f * cc;
        float sig  = 1.0f / (1.0f + __expf(-dist));
        float e    = ct - sig - ms;
        local      = fmaf(e, e, local);
      }
    }
  }

  // deterministic block tree-reduce -> one partial per tile
  red[threadIdx.x] = local;
  __syncthreads();
  for (int s = 128; s > 0; s >>= 1) {
    if (threadIdx.x < s) red[threadIdx.x] += red[threadIdx.x + s];
    __syncthreads();
  }
  if (threadIdx.x == 0) partials[blockIdx.y * gridDim.x + blockIdx.x] = red[0];
}

// fixed-order final reduction -> scalar loss
__global__ void final_reduce_kernel(const float* __restrict__ partials,
                                    float* __restrict__ out) {
  __shared__ float sm[256];
  int tid = threadIdx.x;
  float s = 0.0f;
  for (int i = tid; i < NPART; i += 256) s += partials[i];
  sm[tid] = s;
  __syncthreads();
  for (int st = 128; st > 0; st >>= 1) {
    if (tid < st) sm[tid] += sm[tid + st];
    __syncthreads();
  }
  if (tid == 0) out[0] = sm[0] * (1.0f / ((float)N_ROWS * (float)N_ROWS));
}

extern "C" void kernel_launch(void* const* d_in, const int* in_sizes, int n_in,
                              void* d_out, int out_size, void* d_ws, size_t ws_size,
                              hipStream_t stream) {
  // setup_inputs order: input_img (unused), input_text, caption, labels (unused)
  const float* input_text = (const float*)d_in[1];
  const float* caption    = (const float*)d_in[2];

  char* ws = (char*)d_ws;
  // workspace layout (~6.34 MB total)
  __bf16* t_bf     = (__bf16*)(ws);                              // 8192*256*2 = 4 MB
  __bf16* c_bf     = (__bf16*)(ws + 4194304);                    // 8192*128*2 = 2 MB
  float*  sq       = (float*) (ws + 6291456);                    // 32 KB
  float*  msim     = (float*) (ws + 6291456 + 32768);            // 4 B (padded)
  float*  partials = (float*) (ws + 6291456 + 32768 + 64);       // 16 KB

  prep_text_kernel<<<N_ROWS, D_TEXT, 0, stream>>>(input_text, t_bf);
  prep_cap_kernel <<<N_ROWS, D_CAP,  0, stream>>>(caption, c_bf, sq);
  mean_sim_kernel <<<1,      D_TEXT, 0, stream>>>(t_bf, msim);

  dim3 grid(GRID_T, GRID_T);
  fused_tile_kernel<<<grid, 256, 0, stream>>>(t_bf, c_bf, sq, msim, partials);
  final_reduce_kernel<<<1, 256, 0, stream>>>(partials, (float*)d_out);
}
